// InteractionGNN_82961588289779
// MI455X (gfx1250) — matrix operand order repaired and checked by reading, not compiled, not run on hardware
//
#include <hip/hip_runtime.h>

#define NN     100000
#define EE     600000
#define DIMN   128
#define HID    256
#define KE     257
#define KEP    288
#define ETILE  32
#define NLAYER 2

typedef __attribute__((ext_vector_type(16))) _Float16 v16h;
typedef __attribute__((ext_vector_type(8)))  float    v8f;
typedef __attribute__((ext_vector_type(4)))  int      vi4;

union FragU { v16h h; vi4 v[2]; };

#define WMMA16(A,B,C) __builtin_amdgcn_wmma_f32_16x16x32_f16(false,(A),false,(B),(short)0,(C),false,false)

// A-matrix 16x32 f16 fragment (row m, per ISA layout):
// VGPR0-3: k = kb + half*8 + 0..7 ; VGPR4-7: k = kb + 16 + half*8 + 0..7
__device__ __forceinline__ v16h loadA16(const _Float16* base, int strideH, int m, int half, int kb) {
  FragU f;
  const _Float16* p = base + m * strideH + kb + half * 8;
  f.v[0] = *(const vi4*)p;
  f.v[1] = *(const vi4*)(p + 16);
  return f.h;
}

// B-matrix 32x16 f16 fragment from transposed weights wT[n][k] (stride Kpad):
// lane n: VGPR0-3: k = kb + half*8 + 0..7 ; VGPR4-7: k = kb + 16 + half*8 + 0..7
__device__ __forceinline__ v16h loadB16(const _Float16* wT, int Kpad, int n, int half, int kb) {
  FragU f;
  const _Float16* p = wT + (long long)n * Kpad + kb + half * 8;
  f.v[0] = *(const vi4*)p;
  f.v[1] = *(const vi4*)(p + 16);
  return f.h;
}

__device__ __forceinline__ float gelu_exact(float x) {
  return 0.5f * x * (1.0f + erff(x * 0.70710678118654752f));
}

__global__ __launch_bounds__(256) void zero_f32(float* p, int n) {
  int i = blockIdx.x * 256 + threadIdx.x;
  if (i < n) p[i] = 0.0f;
}

// dst[n*Kpad + k] = (k < K) ? (f16)src[k*Nout + n] : 0   (transpose + f32->f16)
__global__ __launch_bounds__(256) void tcvt(const float* __restrict__ src, _Float16* __restrict__ dst,
                                            int K, int Kpad, int Nout) {
  int idx = blockIdx.x * 256 + threadIdx.x;
  if (idx >= Nout * Kpad) return;
  int n = idx / Kpad, k = idx - n * Kpad;
  dst[idx] = (k < K) ? (_Float16)src[(long long)k * Nout + n] : (_Float16)0.0f;
}

// 32 edges per block, 256 threads (8 waves). Each wave: 2 N-tiles x 2 M-tiles,
// reusing B fragments across both M tiles (halves L2 weight traffic).
__global__ __launch_bounds__(256) void edge_kernel(
    const float* __restrict__ h, const long long* __restrict__ eidx,
    const float* __restrict__ eattr, const _Float16* __restrict__ w1T,
    const float* __restrict__ b1, const float* __restrict__ w2, const float* __restrict__ b2,
    float* __restrict__ agg, float* __restrict__ den)
{
  __shared__ _Float16 sX[ETILE * KEP];    // [h_src | h_dst | attr | pad] f16 A panel (18 KB)
  __shared__ _Float16 sG1[ETILE * HID];   // gelu(x@w1+b1) f16                        (16 KB)
  __shared__ float    sHs[ETILE * DIMN];  // h_src f32 (for message)                  (16 KB)
  __shared__ float    sRed[256];
  __shared__ float    sGate[ETILE];
  __shared__ int      sDst[ETILE];

  const int t    = threadIdx.x;
  const int e    = t >> 3, c = t & 7;     // 8 threads per edge, 16 floats each
  const int lane = t & 31, wv = t >> 5;
  const int ln   = lane & 15, hf = lane >> 4;
  const long long e0 = (long long)blockIdx.x * ETILE;

  int srcIdx = 0;
  if (t < ETILE) {
    long long eg = e0 + t; if (eg >= EE) eg = EE - 1;
    srcIdx  = (int)eidx[eg * 2 + 0];
    sDst[t] = (int)eidx[eg * 2 + 1];
    sX[t * KEP + 256] = (_Float16)eattr[eg];
    for (int k = 257; k < KEP; ++k) sX[t * KEP + k] = (_Float16)0.0f;
  }
  // broadcast src index to the 8 threads of each edge via LDS (reuse sRed)
  if (t < ETILE) sRed[t] = __int_as_float(srcIdx);
  __syncthreads();
  {
    int sidx = __float_as_int(sRed[e]);
    const float* ps = h + (long long)sidx    * DIMN + c * 16;
    const float* pd = h + (long long)sDst[e] * DIMN + c * 16;
    #pragma unroll
    for (int j = 0; j < 16; ++j) {
      float vs = ps[j], vd = pd[j];
      sHs[e * DIMN + c * 16 + j]      = vs;
      sX[e * KEP + c * 16 + j]        = (_Float16)vs;
      sX[e * KEP + 128 + c * 16 + j]  = (_Float16)vd;
    }
  }
  __syncthreads();

  // GEMM1: (32 x 288) @ (288 x 256); wave wv owns n-tiles wv*32, wv*32+16 and both M tiles
  {
    v8f a00 = {}; v8f a01 = {}; v8f a10 = {}; v8f a11 = {};
    const int nb0 = wv * 32, nb1 = wv * 32 + 16;
    #pragma unroll
    for (int kt = 0; kt < KEP / 32; ++kt) {
      const int kb = kt * 32;
      v16h A0 = loadA16(sX,            KEP, ln, hf, kb);
      v16h A1 = loadA16(sX + 16 * KEP, KEP, ln, hf, kb);
      v16h B0 = loadB16(w1T, KEP, nb0 + ln, hf, kb);
      v16h B1 = loadB16(w1T, KEP, nb1 + ln, hf, kb);
      a00 = WMMA16(A0, B0, a00);
      a10 = WMMA16(A1, B0, a10);
      a01 = WMMA16(A0, B1, a01);
      a11 = WMMA16(A1, B1, a11);
    }
    const float bb0 = b1[nb0 + ln], bb1 = b1[nb1 + ln];
    #pragma unroll
    for (int r = 0; r < 8; ++r) {
      int mrow = r + 8 * hf;
      sG1[mrow * HID + nb0 + ln]        = (_Float16)gelu_exact(a00[r] + bb0);
      sG1[mrow * HID + nb1 + ln]        = (_Float16)gelu_exact(a01[r] + bb1);
      sG1[(16 + mrow) * HID + nb0 + ln] = (_Float16)gelu_exact(a10[r] + bb0);
      sG1[(16 + mrow) * HID + nb1 + ln] = (_Float16)gelu_exact(a11[r] + bb1);
    }
  }
  __syncthreads();

  // gate = sigmoid(g1 @ w2 + b2): 8 threads/edge partial dot (32 elems each) + reduce
  {
    float p = 0.0f;
    #pragma unroll
    for (int j = 0; j < 32; ++j)
      p += (float)sG1[e * HID + c * 32 + j] * w2[c * 32 + j];
    sRed[t] = p;
  }
  __syncthreads();
  if (t < ETILE) {
    float s = b2[0];
    #pragma unroll
    for (int j = 0; j < 8; ++j) s += sRed[t * 8 + j];
    sGate[t] = 1.0f / (1.0f + __expf(-s));
  }
  __syncthreads();

  // scatter msg = h_src * gate into agg[dst], gate into den[dst]
  if (e0 + e < EE) {
    float g = sGate[e];
    int   d = sDst[e];
    float* pa = agg + (long long)d * DIMN + c * 16;
    #pragma unroll
    for (int j = 0; j < 16; ++j)
      atomicAdd(pa + j, sHs[e * DIMN + c * 16 + j] * g);
    if (c == 0) atomicAdd(den + d, g);
  }
}

__global__ __launch_bounds__(256) void node_kernel(
    const float* __restrict__ h, const float* __restrict__ agg, const float* __restrict__ den,
    const _Float16* __restrict__ w1T, const float* __restrict__ b1,
    const _Float16* __restrict__ w2T, const float* __restrict__ b2,
    const float* __restrict__ lng, const float* __restrict__ lnb,
    float* __restrict__ hout)
{
  __shared__ _Float16 sU[16 * 256];   // [h | agg_norm] f16
  __shared__ _Float16 sG[16 * 256];   // gelu(u@w1+b1) f16
  __shared__ float    sH[16 * DIMN];  // h f32 (residual)
  __shared__ float    sF[16 * DIMN];  // h + upd f32
  __shared__ float    sMu[16], sRs[16];

  const int t    = threadIdx.x;
  const int e    = t >> 4, c = t & 15;
  const int lane = t & 31, wv = t >> 5;
  const int ln   = lane & 15, hf = lane >> 4;
  const long long row = (long long)blockIdx.x * 16 + e;

  {
    const float* ph = h   + row * DIMN + c * 8;
    const float* pa = agg + row * DIMN + c * 8;
    float inv = 1.0f / fmaxf(den[row], 1e-6f);
    #pragma unroll
    for (int j = 0; j < 8; ++j) {
      float hv = ph[j];
      sH[e * DIMN + c * 8 + j]       = hv;
      sU[e * 256 + c * 8 + j]        = (_Float16)hv;
      sU[e * 256 + 128 + c * 8 + j]  = (_Float16)(pa[j] * inv);
    }
  }
  __syncthreads();

  // GEMM1: (16 x 256) @ (256 x 256)
  {
    v8f a0 = {}; v8f a1 = {};
    const int nb0 = wv * 32, nb1 = wv * 32 + 16;
    #pragma unroll
    for (int kt = 0; kt < 8; ++kt) {
      const int kb = kt * 32;
      v16h A  = loadA16(sU, 256, ln, hf, kb);
      v16h B0 = loadB16(w1T, 256, nb0 + ln, hf, kb);
      v16h B1 = loadB16(w1T, 256, nb1 + ln, hf, kb);
      a0 = WMMA16(A, B0, a0);
      a1 = WMMA16(A, B1, a1);
    }
    #pragma unroll
    for (int r = 0; r < 8; ++r) {
      int mrow = r + 8 * hf;
      sG[mrow * 256 + nb0 + ln] = (_Float16)gelu_exact(a0[r] + b1[nb0 + ln]);
      sG[mrow * 256 + nb1 + ln] = (_Float16)gelu_exact(a1[r] + b1[nb1 + ln]);
    }
  }
  __syncthreads();

  // GEMM2: (16 x 256) @ (256 x 128); wave wv owns n-tile wv*16
  {
    v8f acc = {};
    const int nb = wv * 16;
    #pragma unroll
    for (int kt = 0; kt < 8; ++kt) {
      const int kb = kt * 32;
      v16h A = loadA16(sG, 256, ln, hf, kb);
      v16h B = loadB16(w2T, 256, nb + ln, hf, kb);
      acc = WMMA16(A, B, acc);
    }
    #pragma unroll
    for (int r = 0; r < 8; ++r) {
      int mrow = r + 8 * hf;
      sF[mrow * DIMN + nb + ln] = acc[r] + b2[nb + ln] + sH[mrow * DIMN + nb + ln];
    }
  }
  __syncthreads();

  if (t < 16) {
    float s = 0.0f, s2 = 0.0f;
    for (int j = 0; j < DIMN; ++j) { float v = sF[t * DIMN + j]; s += v; s2 += v * v; }
    float mu  = s * (1.0f / DIMN);
    float var = s2 * (1.0f / DIMN) - mu * mu;
    sMu[t] = mu;
    sRs[t] = rsqrtf(var + 1e-5f);
  }
  __syncthreads();
  {
    float mu = sMu[e], rs = sRs[e];
    float* po = hout + row * DIMN + c * 8;
    #pragma unroll
    for (int j = 0; j < 8; ++j) {
      int d = c * 8 + j;
      po[j] = (sF[e * DIMN + d] - mu) * rs * lng[d] + lnb[d];
    }
  }
}

extern "C" void kernel_launch(void* const* d_in, const int* in_sizes, int n_in,
                              void* d_out, int out_size, void* d_ws, size_t ws_size,
                              hipStream_t stream) {
  const float*     h0    = (const float*)d_in[0];
  const long long* eidx  = (const long long*)d_in[1];
  const float*     eattr = (const float*)d_in[2];
  const float*     ew1   = (const float*)d_in[3];
  const float*     eb1   = (const float*)d_in[4];
  const float*     ew2   = (const float*)d_in[5];
  const float*     eb2   = (const float*)d_in[6];
  const float*     nw1   = (const float*)d_in[7];
  const float*     nb1   = (const float*)d_in[8];
  const float*     nw2   = (const float*)d_in[9];
  const float*     nb2   = (const float*)d_in[10];
  const float*     lng   = (const float*)d_in[11];
  const float*     lnb   = (const float*)d_in[12];

  size_t off = 0;
  auto alloc = [&](size_t bytes) -> char* {
    char* p = (char*)d_ws + off;
    off = (off + bytes + 255) & ~(size_t)255;
    return p;
  };
  float*     hbuf  = (float*)alloc((size_t)NN * DIMN * 4);
  float*     agg   = (float*)alloc((size_t)NN * DIMN * 4);
  float*     den   = (float*)alloc((size_t)NN * 4);
  _Float16*  w1T   = (_Float16*)alloc((size_t)HID * KEP * 2);
  _Float16*  nw1T  = (_Float16*)alloc((size_t)256 * 256 * 2);
  _Float16*  nw2T  = (_Float16*)alloc((size_t)128 * 256 * 2);
  (void)ws_size; (void)in_sizes; (void)n_in; (void)out_size;

  for (int L = 0; L < NLAYER; ++L) {
    const float* hin  = (L == 0) ? h0 : hbuf;
    float*       hout = (L == NLAYER - 1) ? (float*)d_out : hbuf;

    zero_f32<<<(NN * DIMN + 255) / 256, 256, 0, stream>>>(agg, NN * DIMN);
    zero_f32<<<(NN + 255) / 256, 256, 0, stream>>>(den, NN);

    tcvt<<<(HID * KEP + 255) / 256, 256, 0, stream>>>(ew1 + (size_t)L * KE * HID, w1T, KE, KEP, HID);
    tcvt<<<(256 * 256 + 255) / 256, 256, 0, stream>>>(nw1 + (size_t)L * 256 * 256, nw1T, 256, 256, 256);
    tcvt<<<(128 * 256 + 255) / 256, 256, 0, stream>>>(nw2 + (size_t)L * 256 * 128, nw2T, 256, 256, 128);

    edge_kernel<<<EE / ETILE, 256, 0, stream>>>(hin, eidx, eattr, w1T,
                                                eb1 + (size_t)L * HID, ew2 + (size_t)L * HID,
                                                eb2 + L, agg, den);

    node_kernel<<<NN / 16, 256, 0, stream>>>(hin, agg, den,
                                             nw1T, nb1 + (size_t)L * 256,
                                             nw2T, nb2 + (size_t)L * DIMN,
                                             lng + (size_t)L * DIMN, lnb + (size_t)L * DIMN,
                                             hout);
  }
}
